// DFlashAttention_58531814310482
// MI455X (gfx1250) — matrix operand order repaired and checked by reading, not compiled, not run on hardware
//
#include <hip/hip_runtime.h>

// ---------------------------------------------------------------------------
// Types for CDNA5 WMMA (gfx1250, wave32)
// ---------------------------------------------------------------------------
typedef __bf16 bf16_t;
typedef __attribute__((ext_vector_type(16))) __bf16 bvec16;
typedef __attribute__((ext_vector_type(8)))  __bf16 bvec8;
typedef __attribute__((ext_vector_type(8)))  float  fvec8;
typedef __attribute__((ext_vector_type(4)))  unsigned int u32x4;
typedef __attribute__((ext_vector_type(8)))  int i32x8;
typedef __attribute__((ext_vector_type(4)))  int i32x4;

#define WMMA_BF16(a, b, c) \
  __builtin_amdgcn_wmma_f32_16x16x32_bf16(false, (a), false, (b), (short)0, (c), false, false)

// A-fragment (16x32 bf16, MxK): lane l holds row M=l%16.
// lanes 0-15:  elements K = {d0..d0+15}
// lanes 16-31: elements K = {d0+16..d0+31}
union AFrag {
  bvec16 v;
  struct { bvec8 lo; bvec8 hi; } h;
};

static __device__ __forceinline__ bvec16 load_a_frag(const bf16_t* row, int d0, int hi) {
  AFrag u;
  u.h.lo = *(const bvec8*)(row + d0 + hi * 16);
  u.h.hi = *(const bvec8*)(row + d0 + hi * 16 + 8);
  return u.v;
}

// ---------------------------------------------------------------------------
// CDNA5 async global -> LDS copy (16B per lane, ASYNCcnt-tracked)
// ---------------------------------------------------------------------------
static __device__ __forceinline__ uint32_t lds_off32(const void* p) {
  // Flat shared-aperture address: low 32 bits are the LDS byte offset.
  return (uint32_t)(uintptr_t)p;
}

static __device__ __forceinline__ void async_copy16(uint32_t lds_dst, const void* gsrc) {
  asm volatile("global_load_async_to_lds_b128 %0, %1, off"
               :: "v"(lds_dst), "v"((uint64_t)(uintptr_t)gsrc)
               : "memory");
}

#define WAIT_ASYNC0() asm volatile("s_wait_asynccnt 0" ::: "memory")
#define WAIT_DS0()    asm volatile("s_wait_dscnt 0" ::: "memory")

// ---------------------------------------------------------------------------
// Tensor Data Mover: 2D tile load (2-byte elements) into LDS, row-major
// [tile_d1][tile_d0]. D# bitfields per CDNA5 ISA.
// Issued once per wave (EXEC ignored); tracked by TENSORcnt.
// amdgpu-toolchain (clang-23) 6-arg builtin form.
// ---------------------------------------------------------------------------
static __device__ __forceinline__ void tdm_load_2d(uint32_t lds_addr, const void* gaddr,
                                                   uint32_t tensor_d0, uint32_t tensor_d1,
                                                   uint32_t tile_d0, uint32_t tile_d1,
                                                   uint32_t stride_d0 /* elements */) {
  const uint64_t ga = (uint64_t)(uintptr_t)gaddr;
  u32x4 g0;
  g0[0] = 1u;                                   // count=1, user descriptor
  g0[1] = lds_addr;                             // LDS byte address
  g0[2] = (uint32_t)ga;                         // global_addr[31:0]
  g0[3] = (uint32_t)(ga >> 32) | (2u << 30);    // global_addr[56:32] | type=2 (image)
  i32x8 g1;
  g1[0] = 1 << 16;                              // workgroup_mask=0 | data_size=1 (2B)
  g1[1] = (int)((tensor_d0 & 0xFFFFu) << 16);   // tensor_dim0[15:0] (atomic_barrier=0)
  g1[2] = (int)((tensor_d0 >> 16) | ((tensor_d1 & 0xFFFFu) << 16));  // d0[31:16]|d1[15:0]
  g1[3] = (int)((tensor_d1 >> 16) | (tile_d0 << 16));                // d1[31:16]|tile_d0
  g1[4] = (int)(tile_d1 & 0xFFFFu);             // tile_dim1 | tile_dim2=0
  g1[5] = (int)stride_d0;                       // tensor_dim0_stride[31:0]
  g1[6] = 0;                                    // stride[47:32]=0 | dim1_stride lo
  g1[7] = 0;
  i32x4 gz4 = {0, 0, 0, 0};                     // groups 2/3 unused (2D)
  i32x8 gz8 = {0, 0, 0, 0, 0, 0, 0, 0};
  __builtin_amdgcn_tensor_load_to_lds(g0, g1, gz4, gz4, gz8, 0);
}

// ---------------------------------------------------------------------------
// Elementwise f32 -> bf16 convert
// ---------------------------------------------------------------------------
__global__ void cvt_f32_bf16(const float* __restrict__ in, bf16_t* __restrict__ out, size_t n) {
  size_t i = (size_t)blockIdx.x * blockDim.x + threadIdx.x;
  size_t stride = (size_t)gridDim.x * blockDim.x;
  for (; i < n; i += stride) out[i] = (bf16_t)in[i];
}

// ---------------------------------------------------------------------------
// Transpose-convert: in f32 [R][C] -> out bf16 [C][R]   (R,C multiples of 32)
// ---------------------------------------------------------------------------
__global__ void transpose_cvt(const float* __restrict__ in, bf16_t* __restrict__ out,
                              int R, int C) {
  __shared__ float tile[32][33];
  const int c0 = blockIdx.x * 32, r0 = blockIdx.y * 32;
  const int tx = threadIdx.x, ty = threadIdx.y;   // block (32,8)
#pragma unroll
  for (int i = 0; i < 4; ++i)
    tile[ty + i * 8][tx] = in[(size_t)(r0 + ty + i * 8) * C + c0 + tx];
  __syncthreads();
#pragma unroll
  for (int i = 0; i < 4; ++i)
    out[(size_t)(c0 + ty + i * 8) * R + r0 + tx] = (bf16_t)tile[tx][ty + i * 8];
}

// ---------------------------------------------------------------------------
// bf16 GEMM: C[M][Nout] (f32) = A[M][Kd] (bf16) * BT[Nout][Kd]^T (bf16)
// Wave computes a 16x128 tile; 8 waves/block -> 128x128 per block.
// B tile (128 x 32 per k-step) is staged in LDS by the Tensor Data Mover,
// triple-buffered (depth-2 prefetch), wave 0 issues + waits TENSORcnt.
// B fragments are software-pipelined through registers so LDS loads for
// fragment nt+1 overlap the WMMA on fragment nt.
// ---------------------------------------------------------------------------
__global__ __launch_bounds__(256) void gemm_bf16(const bf16_t* __restrict__ A,
                                                 const bf16_t* __restrict__ BT,
                                                 float* __restrict__ C,
                                                 int M, int Nout, int Kd) {
  __shared__ bf16_t bls[3][128 * 32];  // 8 KB per buffer
  const int tid  = threadIdx.x;
  const int lane = tid & 31;
  const int w    = tid >> 5;
  const int l16  = lane & 15;
  const int hi   = lane >> 4;
  const int m0   = blockIdx.x * 128 + w * 16;
  const int n0   = blockIdx.y * 128;

  fvec8 acc[8];
#pragma unroll
  for (int t = 0; t < 8; ++t)
#pragma unroll
    for (int e = 0; e < 8; ++e) acc[t][e] = 0.0f;

  const bf16_t* arow = A + (size_t)(m0 + l16) * Kd;
  const bf16_t* bbase = BT + (size_t)n0 * Kd;

  // TDM fill of one 128x32 B tile into bls[buf]
  auto issue_b = [&](int buf, int d0) {
    tdm_load_2d(lds_off32(&bls[buf][0]), bbase + d0,
                (uint32_t)Kd, 128u, 32u, 128u, (uint32_t)Kd);
  };

  const int steps = Kd / 32;
  if (w == 0) {
    issue_b(0, 0);
    issue_b(1, 32);
  }
  int cur = 0;
  for (int it = 0; it < steps; ++it) {
    if (w == 0) __builtin_amdgcn_s_wait_tensorcnt((short)1);  // tile `it` landed
    __syncthreads();
    const int d0 = it * 32;
    bvec16 a = load_a_frag(arow, d0, hi);
    const bf16_t* bp = &bls[cur][l16 * 32 + hi * 16];
    bvec16 b = *(const bvec16*)bp;
#pragma unroll
    for (int nt = 0; nt < 8; ++nt) {
      bvec16 bn;
      if (nt < 7) bn = *(const bvec16*)(bp + (nt + 1) * 16 * 32);
      acc[nt] = WMMA_BF16(a, b, acc[nt]);
      b = bn;
    }
    if (w == 0 && it + 2 < steps) {
      int nb = cur + 2;
      if (nb >= 3) nb -= 3;
      issue_b(nb, d0 + 64);
    }
    cur = (cur == 2) ? 0 : cur + 1;
  }
  // C layout: VGPR r -> row m0 + r + 8*hi, col = n0 + nt*16 + l16
#pragma unroll
  for (int nt = 0; nt < 8; ++nt)
#pragma unroll
    for (int r = 0; r < 8; ++r)
      C[(size_t)(m0 + hi * 8 + r) * Nout + n0 + nt * 16 + l16] = acc[nt][r];
}

// ---------------------------------------------------------------------------
// RMSNorm + RoPE for Q  (one wave per (t, head) row; 4 heads per block)
// qf: [Tq][N*H] f32 ; qb: [N][Tq][H] bf16, pre-scaled by 1/sqrt(H)
// ---------------------------------------------------------------------------
__global__ void rope_q_kernel(const float* __restrict__ qf, const float* __restrict__ qscale,
                              const int* __restrict__ npos, bf16_t* __restrict__ qb, int Tq) {
  const int t = blockIdx.x, n = blockIdx.y * 4 + threadIdx.y, lane = threadIdx.x;
  const float* row = qf + (size_t)t * 2048 + n * 128;
  float2 x1 = *(const float2*)(row + lane * 2);
  float2 x2 = *(const float2*)(row + 64 + lane * 2);
  float ss = x1.x * x1.x + x1.y * x1.y + x2.x * x2.x + x2.y * x2.y;
#pragma unroll
  for (int m = 1; m < 32; m <<= 1) ss += __shfl_xor(ss, m, 32);
  const float rms = rsqrtf(ss * (1.0f / 128.0f) + 1e-6f);
  const float pos = (float)npos[t];
  const float sc  = 0.08838834764831845f;  // 1/sqrt(128), folded attn scale
  bf16_t* orow = qb + ((size_t)n * Tq + t) * 128;
#pragma unroll
  for (int j = 0; j < 2; ++j) {
    const int i = lane * 2 + j;
    const float xa = j ? x1.y : x1.x;
    const float xb = j ? x2.y : x2.x;
    const float n1 = xa * rms * qscale[i];
    const float n2 = xb * rms * qscale[i + 64];
    const float freq = __expf(-(float)(2 * i) * (1.0f / 128.0f) * 13.815510557964274f);
    float sn, cs;
    __sincosf(pos * freq, &sn, &cs);
    orow[i]      = (bf16_t)((n1 * cs - n2 * sn) * sc);
    orow[i + 64] = (bf16_t)((n2 * cs + n1 * sn) * sc);
  }
}

// ---------------------------------------------------------------------------
// RMSNorm + RoPE for K  (one wave per (s, kv-head); 4 heads per block)
// kf: [S][KH*H] f32 ; kb: [KH][S][H] bf16
// ---------------------------------------------------------------------------
__global__ void rope_k_kernel(const float* __restrict__ kf, const float* __restrict__ kscale,
                              const int* __restrict__ cpos, const int* __restrict__ npos,
                              bf16_t* __restrict__ kb, int S, int Tc) {
  const int s = blockIdx.x, kh = blockIdx.y * 4 + threadIdx.y, lane = threadIdx.x;
  const float* row = kf + (size_t)s * 1024 + kh * 128;
  float2 x1 = *(const float2*)(row + lane * 2);
  float2 x2 = *(const float2*)(row + 64 + lane * 2);
  float ss = x1.x * x1.x + x1.y * x1.y + x2.x * x2.x + x2.y * x2.y;
#pragma unroll
  for (int m = 1; m < 32; m <<= 1) ss += __shfl_xor(ss, m, 32);
  const float rms = rsqrtf(ss * (1.0f / 128.0f) + 1e-6f);
  const float pos = (float)((s < Tc) ? cpos[s] : npos[s - Tc]);
  bf16_t* orow = kb + ((size_t)kh * S + s) * 128;
#pragma unroll
  for (int j = 0; j < 2; ++j) {
    const int i = lane * 2 + j;
    const float xa = j ? x1.y : x1.x;
    const float xb = j ? x2.y : x2.x;
    const float n1 = xa * rms * kscale[i];
    const float n2 = xb * rms * kscale[i + 64];
    const float freq = __expf(-(float)(2 * i) * (1.0f / 128.0f) * 13.815510557964274f);
    float sn, cs;
    __sincosf(pos * freq, &sn, &cs);
    orow[i]      = (bf16_t)(n1 * cs - n2 * sn);
    orow[i + 64] = (bf16_t)(n2 * cs + n1 * sn);
  }
}

// ---------------------------------------------------------------------------
// V transpose-convert: vf f32 [S][KH*H] -> vT bf16 [KH][H][S]
// ---------------------------------------------------------------------------
__global__ void vT_kernel(const float* __restrict__ vf, bf16_t* __restrict__ vT, int S) {
  __shared__ float tile[32][33];
  const int kh = blockIdx.z;
  const int s0 = blockIdx.x * 32, h0 = blockIdx.y * 32;
  const int tx = threadIdx.x, ty = threadIdx.y;  // block (32,8)
#pragma unroll
  for (int i = 0; i < 4; ++i)
    tile[ty + i * 8][tx] = vf[(size_t)(s0 + ty + i * 8) * 1024 + kh * 128 + h0 + tx];
  __syncthreads();
#pragma unroll
  for (int i = 0; i < 4; ++i)
    vT[((size_t)kh * 128 + h0 + ty + i * 8) * S + s0 + tx] = (bf16_t)tile[tx][ty + i * 8];
}

// ---------------------------------------------------------------------------
// Flash attention: per-wave 16-query tile for one head; online softmax.
// K/V s-tiles (32 wide) are shared by all 4 waves of the block and staged in
// LDS with double-buffered per-lane async global->LDS copies (ASYNCcnt).
// qb: [N][Tq][H] bf16 (scaled), kb: [KH][S][H] bf16, vT: [KH][H][S] bf16
// Ob: [Tq][N*H] bf16
// ---------------------------------------------------------------------------
#define ATT_WAVES 4
__global__ __launch_bounds__(32 * ATT_WAVES) void attn_fwd(
    const bf16_t* __restrict__ qb, const bf16_t* __restrict__ kb,
    const bf16_t* __restrict__ vT, bf16_t* __restrict__ Ob, int Tq, int S) {
  __shared__ bf16_t kls[2][32 * 128];          // 8 KB per buffer
  __shared__ bf16_t vls[2][128 * 32];          // 8 KB per buffer
  __shared__ bf16_t plds[ATT_WAVES][16 * 32];  // per-wave P staging
  const int tid  = threadIdx.x;
  const int lane = tid & 31;
  const int w    = tid >> 5;
  const int l16  = lane & 15;
  const int hi   = lane >> 4;
  const int n    = blockIdx.y;
  const int kh   = n >> 1;  // GQA: groups = N/K = 2
  const int m0   = (blockIdx.x * ATT_WAVES + w) * 16;

  const bf16_t* kbase = kb + (size_t)kh * S * 128;
  const bf16_t* vbase = vT + (size_t)kh * 128 * S;

  // Q as 4 A-fragments over H=128 (k-dim chunks of 32)
  const bf16_t* qrow = qb + ((size_t)n * Tq + (m0 + l16)) * 128;
  bvec16 qa[4];
#pragma unroll
  for (int c = 0; c < 4; ++c) qa[c] = load_a_frag(qrow, c * 32, hi);

  fvec8 of[8];
#pragma unroll
  for (int t = 0; t < 8; ++t)
#pragma unroll
    for (int e = 0; e < 8; ++e) of[t][e] = 0.0f;
  float mx[8], sm[8];
#pragma unroll
  for (int r = 0; r < 8; ++r) { mx[r] = -1e30f; sm[r] = 0.0f; }

  // Cooperative async fill of K (32x128) and V (128x32): 512 chunks each,
  // 4 per thread per tensor.
  auto issue_kv = [&](int buf, int s0) {
#pragma unroll
    for (int j = 0; j < 4; ++j) {
      const int c = j * 128 + tid;  // 0..511
      const int krow = c >> 4, kel = (c & 15) * 8;   // 16 chunks per 128-elem row
      async_copy16(lds_off32(&kls[buf][c * 8]),
                   kbase + (size_t)(s0 + krow) * 128 + kel);
      const int vrow = c >> 2, vel = (c & 3) * 8;    // 4 chunks per 32-elem row
      async_copy16(lds_off32(&vls[buf][c * 8]),
                   vbase + (size_t)vrow * S + s0 + vel);
    }
  };

  issue_kv(0, 0);

  const int steps = S / 32;
  for (int it = 0; it < steps; ++it) {
    const int cur = it & 1;
    WAIT_ASYNC0();      // this wave's fill of buf `cur` is complete
    __syncthreads();    // everyone's fill is complete
    if (it + 1 < steps) issue_kv(cur ^ 1, (it + 1) * 32);

    // ---- S tile (16 x 32) = Q * K^T, two 16-col WMMA sub-tiles ----
    fvec8 sa0, sa1;
#pragma unroll
    for (int e = 0; e < 8; ++e) { sa0[e] = 0.0f; sa1[e] = 0.0f; }
    const bf16_t* kp0 = &kls[cur][l16 * 128 + hi * 16];
#pragma unroll
    for (int c = 0; c < 4; ++c) {
      bvec16 b0 = *(const bvec16*)(kp0 + c * 32);
      sa0 = WMMA_BF16(qa[c], b0, sa0);
      bvec16 b1 = *(const bvec16*)(kp0 + 16 * 128 + c * 32);
      sa1 = WMMA_BF16(qa[c], b1, sa1);
    }
    // ---- online softmax; row r lives in lanes sharing the same 16-lane half
#pragma unroll
    for (int r = 0; r < 8; ++r) {
      float v = fmaxf(sa0[r], sa1[r]);
      v = fmaxf(v, __shfl_xor(v, 1, 32));
      v = fmaxf(v, __shfl_xor(v, 2, 32));
      v = fmaxf(v, __shfl_xor(v, 4, 32));
      v = fmaxf(v, __shfl_xor(v, 8, 32));
      const float nm    = fmaxf(mx[r], v);
      const float alpha = __expf(mx[r] - nm);
      const float p0 = __expf(sa0[r] - nm);
      const float p1 = __expf(sa1[r] - nm);
      float rs = p0 + p1;
      rs += __shfl_xor(rs, 1, 32);
      rs += __shfl_xor(rs, 2, 32);
      rs += __shfl_xor(rs, 4, 32);
      rs += __shfl_xor(rs, 8, 32);
      sm[r] = sm[r] * alpha + rs;
      mx[r] = nm;
#pragma unroll
      for (int t = 0; t < 8; ++t) of[t][r] *= alpha;
      const int m = r + hi * 8;
      plds[w][m * 32 + l16]      = (bf16_t)p0;
      plds[w][m * 32 + 16 + l16] = (bf16_t)p1;
    }
    WAIT_DS0();
    // ---- reload P as an A-fragment (16x32) from this wave's LDS region ----
    AFrag up;
    up.h.lo = *(const bvec8*)&plds[w][l16 * 32 + hi * 16];
    up.h.hi = *(const bvec8*)&plds[w][l16 * 32 + hi * 16 + 8];
    const bvec16 pa = up.v;
    // ---- O += P * V over 8 h-tiles of 16 (B frags pipelined in registers)
    const bf16_t* vp = &vls[cur][l16 * 32 + hi * 16];
    bvec16 vb = *(const bvec16*)vp;
#pragma unroll
    for (int t = 0; t < 8; ++t) {
      bvec16 vbn;
      if (t < 7) vbn = *(const bvec16*)(vp + (t + 1) * 16 * 32);
      of[t] = WMMA_BF16(pa, vb, of[t]);
      vb = vbn;
    }
  }
  // ---- normalize and store bf16 [Tq][N*H] ----
#pragma unroll
  for (int t = 0; t < 8; ++t)
#pragma unroll
    for (int r = 0; r < 8; ++r) {
      const float o = of[t][r] / sm[r];
      Ob[(size_t)(m0 + hi * 8 + r) * 2048 + n * 128 + t * 16 + l16] = (bf16_t)o;
    }
}

// ---------------------------------------------------------------------------
// Host orchestration
// ---------------------------------------------------------------------------
extern "C" void kernel_launch(void* const* d_in, const int* in_sizes, int n_in,
                              void* d_out, int out_size, void* d_ws, size_t ws_size,
                              hipStream_t stream) {
  (void)in_sizes; (void)n_in; (void)out_size; (void)ws_size;
  constexpr int D = 2048, N = 16, KH = 8, H = 128, Tq = 2048, Tc = 4096;
  constexpr int S = Tq + Tc;  // 6144

  const float* x_noise = (const float*)d_in[0];
  const float* target  = (const float*)d_in[1];
  const float* Wq = (const float*)d_in[2];
  const float* Wk = (const float*)d_in[3];
  const float* Wv = (const float*)d_in[4];
  const float* Wo = (const float*)d_in[5];
  const float* qs = (const float*)d_in[6];
  const float* ks = (const float*)d_in[7];
  const int* npos = (const int*)d_in[8];
  const int* cpos = (const int*)d_in[9];
  float* out = (float*)d_out;

  char* base = (char*)d_ws;
  size_t off = 0;
  auto take = [&](size_t bytes) -> char* {
    char* r = base + off;
    off += (bytes + 255) & ~(size_t)255;
    return r;
  };
  bf16_t* xb  = (bf16_t*)take((size_t)S * D * 2);        // concat(ctx, noise) bf16
  bf16_t* WqT = (bf16_t*)take((size_t)D * N * H * 2);    // [nh][d]
  bf16_t* WkT = (bf16_t*)take((size_t)D * KH * H * 2);   // [kh*H+h][d]
  bf16_t* WvT = (bf16_t*)take((size_t)D * KH * H * 2);
  bf16_t* WoT = (bf16_t*)take((size_t)D * N * H * 2);    // [d][nh]
  float*  qf  = (float*)take((size_t)Tq * N * H * 4);
  float*  kf  = (float*)take((size_t)S * KH * H * 4);
  float*  vf  = (float*)take((size_t)S * KH * H * 4);
  bf16_t* qb  = (bf16_t*)take((size_t)N * Tq * H * 2);
  bf16_t* kb  = (bf16_t*)take((size_t)KH * S * H * 2);
  bf16_t* vT  = (bf16_t*)take((size_t)KH * H * S * 2);
  bf16_t* Ob  = (bf16_t*)take((size_t)Tq * N * H * 2);

  // 1) activations -> bf16
  cvt_f32_bf16<<<2048, 256, 0, stream>>>(target, xb, (size_t)Tc * D);
  cvt_f32_bf16<<<2048, 256, 0, stream>>>(x_noise, xb + (size_t)Tc * D, (size_t)Tq * D);

  // 2) weights -> transposed bf16 (B-fragment friendly)
  dim3 tb(32, 8);
  transpose_cvt<<<dim3((N * H) / 32, D / 32), tb, 0, stream>>>(Wq, WqT, D, N * H);
  transpose_cvt<<<dim3((KH * H) / 32, D / 32), tb, 0, stream>>>(Wk, WkT, D, KH * H);
  transpose_cvt<<<dim3((KH * H) / 32, D / 32), tb, 0, stream>>>(Wv, WvT, D, KH * H);
  transpose_cvt<<<dim3(D / 32, (N * H) / 32), tb, 0, stream>>>(Wo, WoT, N * H, D);

  // 3) projections (WMMA bf16, f32 accumulate, TDM-staged B)
  gemm_bf16<<<dim3(Tq / 128, (N * H) / 128), 256, 0, stream>>>(xb + (size_t)Tc * D, WqT, qf, Tq, N * H, D);
  gemm_bf16<<<dim3(S / 128, (KH * H) / 128), 256, 0, stream>>>(xb, WkT, kf, S, KH * H, D);
  gemm_bf16<<<dim3(S / 128, (KH * H) / 128), 256, 0, stream>>>(xb, WvT, vf, S, KH * H, D);

  // 4) norm + rope + layout conversion
  rope_q_kernel<<<dim3(Tq, N / 4), dim3(32, 4), 0, stream>>>(qf, qs, npos, qb, Tq);
  rope_k_kernel<<<dim3(S, KH / 4), dim3(32, 4), 0, stream>>>(kf, ks, cpos, npos, kb, S, Tc);
  vT_kernel<<<dim3(S / 32, H / 32, KH), tb, 0, stream>>>(vf, vT, S);

  // 5) flash attention (async-LDS staged K/V)
  attn_fwd<<<dim3(Tq / (16 * ATT_WAVES), N), 32 * ATT_WAVES, 0, stream>>>(qb, kb, vT, Ob, Tq, S);

  // 6) output projection -> d_out (f32)
  gemm_bf16<<<dim3(Tq / 128, D / 128), 256, 0, stream>>>(Ob, WoT, out, Tq, D, N * H);
}
